// HRSBlock_58634893525226
// MI455X (gfx1250) — matrix operand order repaired and checked by reading, not compiled, not run on hardware
//
#include <hip/hip_runtime.h>
#include <hip/hip_bf16.h>

typedef __attribute__((ext_vector_type(16))) _Float16 v16h;
typedef __attribute__((ext_vector_type(8)))  float    v8f;

#define Bn   4
#define Tn   1024
#define Dn   1024
#define Hn   16
#define HDn  64
#define DFFn 2048
#define BTn  (Bn*Tn)

#define GK   32      // WMMA K step (f16)
#define GTM  128     // GEMM block tile M
#define GTN  128     // GEMM block tile N

// ---------------------------------------------------------------- helpers ---

static __device__ __forceinline__ float gelu_tanh(float x){
  float x3 = x*x*x;
  return 0.5f*x*(1.0f + tanhf(0.7978845608028654f*(x + 0.044715f*x3)));
}

// pack 4 floats -> 4 f16, single 8-byte LDS store
static __device__ __forceinline__ void st4h(_Float16* dst, float4 v){
  union { _Float16 h[4]; unsigned long long u; } p;
  p.h[0] = (_Float16)v.x; p.h[1] = (_Float16)v.y;
  p.h[2] = (_Float16)v.z; p.h[3] = (_Float16)v.w;
  *(unsigned long long*)dst = p.u;
}

struct Chunk8  { float f[8];  };
struct Chunk16 { float f[16]; };

static __device__ __forceinline__ Chunk8 ld8(const float* p){
  Chunk8 c;
  float4 a = *(const float4*)p;
  float4 b = *(const float4*)(p + 4);
  c.f[0]=a.x; c.f[1]=a.y; c.f[2]=a.z; c.f[3]=a.w;
  c.f[4]=b.x; c.f[5]=b.y; c.f[6]=b.z; c.f[7]=b.w;
  return c;
}
static __device__ __forceinline__ Chunk16 ld16(const float* p){
  Chunk16 c;
#pragma unroll
  for (int i=0;i<4;++i){
    float4 a = *(const float4*)(p + i*4);
    c.f[i*4+0]=a.x; c.f[i*4+1]=a.y; c.f[i*4+2]=a.z; c.f[i*4+3]=a.w;
  }
  return c;
}

// Build A fragment (16x32 f16) from LDS tile with row stride GK halves.
// Contiguous k order (same as frag_b_lds): lane m = lane&15, g = lane>>4; K(e) = g*16 + e.
static __device__ __forceinline__ v16h frag_a_lds(const _Float16* a, int m0, int lane){
  int m = m0 + (lane & 15);
  int g = (lane >> 4) & 1;
  const unsigned int* a32 = (const unsigned int*)(a + m*GK);
  union { v16h h; unsigned int u[8]; } r;
#pragma unroll
  for (int j = 0; j < 4; ++j) r.u[j]   = a32[g*8 + j];
#pragma unroll
  for (int j = 0; j < 4; ++j) r.u[4+j] = a32[g*8 + 4 + j];
  return r.h;
}

// Build B fragment (32x16 f16) from LDS tile stored n-major: b[n*GK + k].
// Lane: n = lane&15, halves e: K = (lane>>4)*16 + e.
static __device__ __forceinline__ v16h frag_b_lds(const _Float16* bnk, int n0, int lane){
  int n = n0 + (lane & 15);
  int g = (lane >> 4) & 1;
  const unsigned int* b32 = (const unsigned int*)(bnk + n*GK);
  union { v16h h; unsigned int u[8]; } r;
#pragma unroll
  for (int j = 0; j < 8; ++j) r.u[j] = b32[g*8 + j];
  return r.h;
}

static __device__ __forceinline__ v8f wmma_f16(v16h a, v16h b, v8f c){
  return __builtin_amdgcn_wmma_f32_16x16x32_f16(false, a, false, b, (short)0, c, false, false);
}

// ---------------------------------------------------------------- RoPE table -

__global__ void k_rope(float* __restrict__ c, float* __restrict__ s){
  int t = blockIdx.x, i = threadIdx.x;              // block = 32
  float f = (float)t * __expf(-(float)i * (9.210340371976184f/32.0f));
  float sv, cv; sincosf(f, &sv, &cv);
  c[t*32 + i] = cv; s[t*32 + i] = sv;
}

// ---------------------------------------------------------------- LayerNorm --

__global__ __launch_bounds__(256) void k_ln(const float* __restrict__ x,
                                            const float* __restrict__ w,
                                            const float* __restrict__ bb,
                                            float* __restrict__ out){
  __shared__ float red[256];
  int row = blockIdx.x, tid = threadIdx.x;
  const float* xr = x + (size_t)row*Dn + tid*4;
  float4 v = *(const float4*)xr;
  float s = v.x + v.y + v.z + v.w;
  red[tid] = s; __syncthreads();
  for (int st2=128; st2>0; st2>>=1){ if (tid<st2) red[tid]+=red[tid+st2]; __syncthreads(); }
  float mean = red[0] * (1.0f/Dn);
  __syncthreads();
  float dx = v.x-mean, dy = v.y-mean, dz = v.z-mean, dw = v.w-mean;
  red[tid] = dx*dx + dy*dy + dz*dz + dw*dw; __syncthreads();
  for (int st2=128; st2>0; st2>>=1){ if (tid<st2) red[tid]+=red[tid+st2]; __syncthreads(); }
  float rstd = rsqrtf(red[0]*(1.0f/Dn) + 1e-5f);
  float4 wv = *(const float4*)(w  + tid*4);
  float4 bv = *(const float4*)(bb + tid*4);
  float4 o;
  o.x = dx*rstd*wv.x + bv.x; o.y = dy*rstd*wv.y + bv.y;
  o.z = dz*rstd*wv.z + bv.z; o.w = dw*rstd*wv.w + bv.w;
  *(float4*)(out + (size_t)row*Dn + tid*4) = o;
}

// ------------------------------------------------------------ WMMA GEMM ------
// C[z] = epilogue(alpha * A[z] @ op(B[z]) [+ Res[z]])
// A: [M,K] row-major (lda=K). B: normal [K,N] (ldb=N) or transB [N,K] (ldb=K).
// Block tile 128x128, 8 waves, each wave computes 32Mx64N (8 WMMAs / K-step).

struct Stage { float4 a[4]; float4 b[4]; };

static __device__ __forceinline__ Stage stage_load(const float* __restrict__ A,
    const float* __restrict__ Bm, int N, int K, int transB,
    int m0b, int n0b, int k0, int tid){
  Stage s;
#pragma unroll
  for (int i=0;i<4;++i){                       // A tile 128x32: 4 float4/thread
    int idx4 = tid + i*256;
    int r = idx4 >> 3, c4 = idx4 & 7;
    s.a[i] = *(const float4*)(A + (size_t)(m0b + r)*K + k0 + c4*4);
  }
  float4 z; z.x = z.y = z.z = z.w = 0.0f;
#pragma unroll
  for (int i=0;i<4;++i){                       // B tile 128x32: 4 float4/thread
    int idx4 = tid + i*256;
    if (transB){
      int n = idx4 >> 3, c4 = idx4 & 7;        // [N][K]: contiguous along k
      int gn = n0b + n;
      s.b[i] = (gn < N) ? *(const float4*)(Bm + (size_t)gn*K + k0 + c4*4) : z;
    } else {
      int k = idx4 >> 5, n4 = idx4 & 31;       // [K][N]: contiguous along n
      int gn = n0b + n4*4;
      s.b[i] = (gn < N) ? *(const float4*)(Bm + (size_t)(k0 + k)*N + gn) : z;
    }
  }
  return s;
}

static __device__ __forceinline__ void stage_store(const Stage& s,
    _Float16* alds, _Float16* blds, int transB, int tid){
#pragma unroll
  for (int i=0;i<4;++i){
    int idx4 = tid + i*256;
    st4h(alds + idx4*4, s.a[i]);               // contiguous: ds_store_b64
  }
#pragma unroll
  for (int i=0;i<4;++i){
    int idx4 = tid + i*256;
    if (transB){
      int n = idx4 >> 3, c4 = idx4 & 7;
      st4h(blds + n*GK + c4*4, s.b[i]);
    } else {                                   // scatter 4 halves across n rows
      int k = idx4 >> 5, n4 = idx4 & 31;
      float fv[4] = {s.b[i].x, s.b[i].y, s.b[i].z, s.b[i].w};
#pragma unroll
      for (int j=0;j<4;++j) blds[(n4*4 + j)*GK + k] = (_Float16)fv[j];
    }
  }
}

__global__ __launch_bounds__(256) void k_gemm(
    const float* __restrict__ A, const float* __restrict__ Bm,
    const float* __restrict__ Res, float* __restrict__ C,
    int M, int N, int K,
    long long sA, long long sB, long long sC, long long sR,
    float alpha, int transB, int actGelu)
{
  (void)M;
  __shared__ __attribute__((aligned(16))) _Float16 alds[GTM*GK];  // 8 KB
  __shared__ __attribute__((aligned(16))) _Float16 blds[GTN*GK];  // 8 KB
  int z = blockIdx.z;
  A  += (size_t)z * (size_t)sA;
  Bm += (size_t)z * (size_t)sB;
  C  += (size_t)z * (size_t)sC;
  const float* R = Res ? (Res + (size_t)z * (size_t)sR) : nullptr;
  int n0b = blockIdx.x * GTN;
  int m0b = blockIdx.y * GTM;
  int tid = threadIdx.x;
  int wave = tid >> 5, lane = tid & 31;
  int wm = wave & 3, wn = wave >> 2;           // wave tile: 32M x 64N

  v8f acc[2][4];
#pragma unroll
  for (int i=0;i<2;++i)
#pragma unroll
    for (int nt=0;nt<4;++nt)
#pragma unroll
      for (int j=0;j<8;++j) acc[i][nt][j] = 0.0f;

  Stage st = stage_load(A, Bm, N, K, transB, m0b, n0b, 0, tid);

  for (int k0 = 0; k0 < K; k0 += GK){
    __syncthreads();
    stage_store(st, alds, blds, transB, tid);
    __syncthreads();
    if (k0 + GK < K)                 // prefetch next tile into registers,
      st = stage_load(A, Bm, N, K, transB, m0b, n0b, k0 + GK, tid);
    if (k0 + 2*GK < K)               // and hint the one after into L2/WGP$
      __builtin_prefetch(A + (size_t)(m0b + (tid >> 3))*K + k0 + 2*GK, 0, 1);
    v16h af0 = frag_a_lds(alds, wm*32,      lane);
    v16h af1 = frag_a_lds(alds, wm*32 + 16, lane);
#pragma unroll
    for (int nt=0; nt<4; ++nt){
      v16h bf = frag_b_lds(blds, wn*64 + nt*16, lane);
      acc[0][nt] = wmma_f16(af0, bf, acc[0][nt]);
      acc[1][nt] = wmma_f16(af1, bf, acc[1][nt]);
    }
  }

  int g = lane >> 4, nl = lane & 15;
#pragma unroll
  for (int i=0;i<2;++i){
    int mw = m0b + wm*32 + i*16;
#pragma unroll
    for (int nt=0;nt<4;++nt){
      int gn = n0b + wn*64 + nt*16 + nl;
      if (gn < N){
#pragma unroll
        for (int r=0;r<8;++r){
          int gm = mw + g*8 + r;
          float v = acc[i][nt][r]*alpha;
          if (R) v += R[(size_t)gm*N + gn];
          if (actGelu) v = gelu_tanh(v);
          C[(size_t)gm*N + gn] = v;
        }
      }
    }
  }
}

// ----------------------------------------------- backbone flash attention ----
// qkv: (B,T,3,H,HD) fp32. ao: (B,T,D) fp32 (D = H*HD, matching transpose+reshape).
// Grid: (T/128, B*H). Block 256 = 8 waves; wave handles one 16-query tile.

__global__ __launch_bounds__(256) void k_attn(const float* __restrict__ qkv,
                                              const float* __restrict__ rc,
                                              const float* __restrict__ rs,
                                              float* __restrict__ ao){
  __shared__ __attribute__((aligned(16))) _Float16 plds[8][16*GK];   // 8 KB
  __shared__ __attribute__((aligned(16))) _Float16 vlds[8][32*HDn];  // 32 KB
  int tid = threadIdx.x;
  int wave = tid >> 5, lane = tid & 31;
  int g = lane >> 4, nl = lane & 15;
  int bh = blockIdx.y;
  int b = bh >> 4, h = bh & 15;
  int qt = blockIdx.x*8 + wave;
  int tq0 = qt*16;

  // ---- Q fragments (2 x 16x32) with RoPE, pre-scaled by 1/sqrt(HD)=0.125 ----
  v16h qf[2];
  {
    int t = tq0 + nl;
    const float* qr = qkv + ((size_t)(b*Tn + t)*3 + 0)*Dn + h*HDn;
    Chunk8 qa = ld8(qr + g*16);
    Chunk8 qb = ld8(qr + 8 + g*16);
    Chunk8 qc = ld8(qr + 32 + g*16);
    Chunk8 qd = ld8(qr + 40 + g*16);
    Chunk8 c0 = ld8(rc + (size_t)t*32 + g*16);
    Chunk8 c1 = ld8(rc + (size_t)t*32 + 8 + g*16);
    Chunk8 s0c = ld8(rs + (size_t)t*32 + g*16);
    Chunk8 s1c = ld8(rs + (size_t)t*32 + 8 + g*16);
    union { v16h h16; _Float16 f[16]; } u0, u1;
#pragma unroll
    for (int e=0;e<8;++e){
      u0.f[e]   = (_Float16)((qa.f[e]*c0.f[e] - qc.f[e]*s0c.f[e]) * 0.125f);
      u0.f[e+8] = (_Float16)((qb.f[e]*c1.f[e] - qd.f[e]*s1c.f[e]) * 0.125f);
      u1.f[e]   = (_Float16)((qa.f[e]*s0c.f[e] + qc.f[e]*c0.f[e]) * 0.125f);
      u1.f[e+8] = (_Float16)((qb.f[e]*s1c.f[e] + qd.f[e]*c1.f[e]) * 0.125f);
    }
    qf[0] = u0.h16; qf[1] = u1.h16;
  }

  v8f o[4];
  float mrow[8], lrow[8];
#pragma unroll
  for (int j=0;j<4;++j)
#pragma unroll
    for (int r=0;r<8;++r) o[j][r] = 0.0f;
#pragma unroll
  for (int r=0;r<8;++r){ mrow[r] = -1e30f; lrow[r] = 0.0f; }

  int nkb = (qt >> 1) + 1;                 // 32-key blocks up to the diagonal
  for (int kb = 0; kb < nkb; ++kb){
    int kb0 = kb*32;

    // ---- stage this wave's 32x64 V tile into LDS (f16), one key row/lane ----
    {
      int key = kb0 + lane;
      const float* vr = qkv + ((size_t)(b*Tn + key)*3 + 2)*Dn + h*HDn;
      _Float16* vw = &vlds[wave][lane*HDn];
#pragma unroll
      for (int c4=0;c4<16;++c4) st4h(vw + c4*4, *(const float4*)(vr + c4*4));
    }

    // ---- S = Q K^T (rope on K), two 16-key column tiles ----
    v8f s0, s1;
#pragma unroll
    for (int r=0;r<8;++r){ s0[r]=0.f; s1[r]=0.f; }
#pragma unroll
    for (int half = 0; half < 2; ++half){
      int key = kb0 + half*16 + nl;        // B-layout: lane n = key in tile
      const float* kr = qkv + ((size_t)(b*Tn + key)*3 + 1)*Dn + h*HDn;
      Chunk16 x1 = ld16(kr + g*16);        // K(e) = g*16 + e (+32 for frag 1)
      Chunk16 x2 = ld16(kr + 32 + g*16);
      Chunk16 cc = ld16(rc + (size_t)key*32 + g*16);
      Chunk16 ss = ld16(rs + (size_t)key*32 + g*16);
      union { v16h h16; _Float16 f[16]; } u0, u1;
#pragma unroll
      for (int e=0;e<16;++e){
        u0.f[e] = (_Float16)(x1.f[e]*cc.f[e] - x2.f[e]*ss.f[e]);   // dims 0..31
        u1.f[e] = (_Float16)(x1.f[e]*ss.f[e] + x2.f[e]*cc.f[e]);   // dims 32..63
      }
      if (half == 0){ s0 = wmma_f16(qf[0], u0.h16, s0); s0 = wmma_f16(qf[1], u1.h16, s0); }
      else          { s1 = wmma_f16(qf[0], u0.h16, s1); s1 = wmma_f16(qf[1], u1.h16, s1); }
    }

    // ---- online softmax update (rows = g*8+r, 16 lanes per row) ----
    int key0 = kb0 + nl, key1 = kb0 + 16 + nl;
#pragma unroll
    for (int r=0;r<8;++r){
      int qa2 = tq0 + g*8 + r;
      float a0 = (key0 <= qa2) ? s0[r] : -1e30f;
      float a1 = (key1 <= qa2) ? s1[r] : -1e30f;
      float mx = fmaxf(a0, a1);
#pragma unroll
      for (int msk=1; msk<16; msk<<=1) mx = fmaxf(mx, __shfl_xor(mx, msk, 32));
      float mnew = fmaxf(mrow[r], mx);
      float scal = __expf(mrow[r] - mnew);
      float p0 = (key0 <= qa2) ? __expf(s0[r] - mnew) : 0.0f;
      float p1 = (key1 <= qa2) ? __expf(s1[r] - mnew) : 0.0f;
      float ps = p0 + p1;
#pragma unroll
      for (int msk=1; msk<16; msk<<=1) ps += __shfl_xor(ps, msk, 32);
      lrow[r] = lrow[r]*scal + ps;
      mrow[r] = mnew;
#pragma unroll
      for (int j=0;j<4;++j) o[j][r] *= scal;
      plds[wave][(g*8+r)*GK + nl]      = (_Float16)p0;   // C-layout -> LDS
      plds[wave][(g*8+r)*GK + 16 + nl] = (_Float16)p1;
    }

    // ---- PV: P (16x32) as A-fragment, V B-fragments from staged LDS tile ----
    v16h pf = frag_a_lds(plds[wave], 0, lane);
#pragma unroll
    for (int j=0;j<4;++j){
      union { v16h h16; _Float16 f[16]; } u;
      const _Float16* vb = &vlds[wave][(g*16)*HDn + j*16 + nl];
#pragma unroll
      for (int e=0;e<16;++e) u.f[e] = vb[e*HDn];
      o[j] = wmma_f16(pf, u.h16, o[j]);
    }
  }

#pragma unroll
  for (int j=0;j<4;++j)
#pragma unroll
    for (int r=0;r<8;++r){
      int t = tq0 + g*8 + r;
      ao[(size_t)(b*Tn + t)*Dn + h*HDn + j*16 + nl] = o[j][r] / lrow[r];
    }
}

// ---------------------------------------------- tier-2 causal softmax (in-place)

__global__ __launch_bounds__(256) void k_softmax2(float* __restrict__ S){
  __shared__ float red[256];
  int row = blockIdx.x;
  int t = row & (Tn - 1);
  float* sr = S + (size_t)row*Tn;
  int L = t + 1;
  int tid = threadIdx.x;
  float mx = -1e30f;
  for (int j = tid; j < L; j += 256) mx = fmaxf(mx, sr[j]);
  red[tid] = mx; __syncthreads();
  for (int s2 = 128; s2 > 0; s2 >>= 1){ if (tid < s2) red[tid] = fmaxf(red[tid], red[tid+s2]); __syncthreads(); }
  mx = red[0]; __syncthreads();
  float sum = 0.f;
  for (int j = tid; j < L; j += 256) sum += __expf(sr[j]-mx);
  red[tid] = sum; __syncthreads();
  for (int s2 = 128; s2 > 0; s2 >>= 1){ if (tid < s2) red[tid] += red[tid+s2]; __syncthreads(); }
  float inv = 1.0f / red[0];
  for (int j = tid; j < Tn; j += 256) sr[j] = (j < L) ? __expf(sr[j]-mx)*inv : 0.0f;
}

// ---------------------------------------------------------------- router -----

__global__ void k_router(const float* __restrict__ logits, float* __restrict__ rw){
  int row = blockIdx.x*blockDim.x + threadIdx.x;
  if (row >= BTn) return;
  float4 lv = *(const float4*)(logits + (size_t)row*4);
  float l[4] = {lv.x, lv.y, lv.z, lv.w};
  float mx = fmaxf(fmaxf(l[0],l[1]), fmaxf(l[2],l[3]));
  float s = 0.f;
#pragma unroll
  for (int i=0;i<4;++i){ l[i] = __expf(l[i]-mx); s += l[i]; }
  float inv = 1.0f/s;
  float4 ov; ov.x = l[0]*inv; ov.y = l[1]*inv; ov.z = l[2]*inv; ov.w = l[3]*inv;
  *(float4*)(rw + (size_t)row*4) = ov;
}

// --------------------------------------------------- depthwise causal conv ---

__global__ __launch_bounds__(256) void k_conv(const float* __restrict__ h,
                                              const float* __restrict__ w,
                                              float* __restrict__ out){
  int row = blockIdx.x;
  int b = row >> 10, t = row & (Tn-1);
  int d0 = threadIdx.x*4;
  float acc[4] = {0.f,0.f,0.f,0.f};
#pragma unroll
  for (int kk=0; kk<3; ++kk){
    int ts = t + kk - 2;
    if (ts >= 0){
      float4 hv = *(const float4*)(h + (size_t)(b*Tn + ts)*Dn + d0);
      acc[0] += hv.x * w[(d0+0)*3 + kk];
      acc[1] += hv.y * w[(d0+1)*3 + kk];
      acc[2] += hv.z * w[(d0+2)*3 + kk];
      acc[3] += hv.w * w[(d0+3)*3 + kk];
    }
  }
  float4 ov;
  ov.x = gelu_tanh(acc[0]); ov.y = gelu_tanh(acc[1]);
  ov.z = gelu_tanh(acc[2]); ov.w = gelu_tanh(acc[3]);
  *(float4*)(out + (size_t)row*Dn + d0) = ov;
}

// ---------------------------------------------------------------- combine ----

__global__ __launch_bounds__(256) void k_combine(const float* __restrict__ x2,
                                                 const float* __restrict__ conv,
                                                 const float* __restrict__ eo,
                                                 const float* __restrict__ a2,
                                                 const float* __restrict__ rw,
                                                 const float* __restrict__ gate,
                                                 const float* __restrict__ sinks,
                                                 float* __restrict__ out){
  int row = blockIdx.x;
  float4 rv = *(const float4*)(rw + (size_t)row*4);
  float kvs = 1.0f - rv.w*(1.0f - fabsf(sinks[0]));
  int d0 = threadIdx.x*4;
  size_t idx = (size_t)row*Dn + d0;
  float4 cv = *(const float4*)(conv + idx);
  float4 ev = *(const float4*)(eo   + idx);
  float4 av = *(const float4*)(a2   + idx);
  float4 xv = *(const float4*)(x2   + idx);
  float4 gv = *(const float4*)(gate + d0);
  float4 ov;
  ov.x = (xv.x + (cv.x*rv.x + ev.x*rv.y + av.x*rv.z)*gv.x) * kvs;
  ov.y = (xv.y + (cv.y*rv.x + ev.y*rv.y + av.y*rv.z)*gv.y) * kvs;
  ov.z = (xv.z + (cv.z*rv.x + ev.z*rv.y + av.z*rv.z)*gv.z) * kvs;
  ov.w = (xv.w + (cv.w*rv.x + ev.w*rv.y + av.w*rv.z)*gv.w) * kvs;
  *(float4*)(out + idx) = ov;
}

// ---------------------------------------------------------------- launcher ---

extern "C" void kernel_launch(void* const* d_in, const int* in_sizes, int n_in,
                              void* d_out, int out_size, void* d_ws, size_t ws_size,
                              hipStream_t stream){
  (void)in_sizes; (void)n_in; (void)out_size; (void)ws_size;
  const float* x       = (const float*)d_in[0];
  const float* ln1w    = (const float*)d_in[1];
  const float* ln1b    = (const float*)d_in[2];
  const float* qkvw    = (const float*)d_in[3];
  const float* outw    = (const float*)d_in[4];
  const float* ln2w    = (const float*)d_in[5];
  const float* ln2b    = (const float*)d_in[6];
  const float* routerw = (const float*)d_in[7];
  const float* convw   = (const float*)d_in[8];
  const float* ew1     = (const float*)d_in[9];
  const float* ew2     = (const float*)d_in[10];
  const float* atwq    = (const float*)d_in[11];
  const float* atwk    = (const float*)d_in[12];
  const float* atwv    = (const float*)d_in[13];
  const float* atwo    = (const float*)d_in[14];
  const float* gate    = (const float*)d_in[15];
  const float* sinks   = (const float*)d_in[16];

  float* xout  = (float*)d_out;
  float* rwout = xout + (size_t)BTn*Dn;

  // workspace arena (floats), ~151 MB with reuse
  float* w = (float*)d_ws;
  size_t off = 0;
  float* ropec = w + off; off += (size_t)Tn*32;
  float* ropes = w + off; off += (size_t)Tn*32;
  const size_t U = (size_t)BTn*Dn;                 // 4M floats
  float* h1   = w + off; off += U;                 // -> reused as conv_out
  float* big  = w + off; off += 3*U;               // qkv -> emid -> q2/k2/v2 -> a2
  float* ao   = w + off; off += U;                 // -> reused as a2o
  float* x2   = w + off; off += U;
  float* hh   = w + off; off += U;
  float* eout = w + off; off += U;
  float* S2   = w + off; off += (size_t)Bn*Tn*Tn;  // 4M floats

  float* qkv  = big;
  float* conv = h1;
  float* emid = big;
  float* q2 = big; float* k2 = big + U; float* v2 = big + 2*U;
  float* a2 = big;
  float* a2o = ao;
  float* rlog = S2;  // tiny temp before S2 is produced

  // --- backbone ---
  k_rope<<<Tn, 32, 0, stream>>>(ropec, ropes);
  k_ln<<<BTn, 256, 0, stream>>>(x, ln1w, ln1b, h1);
  k_gemm<<<dim3(3*Dn/GTN, BTn/GTM, 1), 256, 0, stream>>>(h1, qkvw, nullptr, qkv,
      BTn, 3*Dn, Dn, 0,0,0,0, 1.0f, 0, 0);
  k_attn<<<dim3(Tn/128, Bn*Hn, 1), 256, 0, stream>>>(qkv, ropec, ropes, ao);
  k_gemm<<<dim3(Dn/GTN, BTn/GTM, 1), 256, 0, stream>>>(ao, outw, x, x2,
      BTn, Dn, Dn, 0,0,0,0, 1.0f, 0, 0);

  // --- router + tiers ---
  k_ln<<<BTn, 256, 0, stream>>>(x2, ln2w, ln2b, hh);
  k_gemm<<<dim3(1, BTn/GTM, 1), 256, 0, stream>>>(hh, routerw, nullptr, rlog,
      BTn, 4, Dn, 0,0,0,0, 1.0f, 0, 0);
  k_router<<<BTn/256, 256, 0, stream>>>(rlog, rwout);
  k_conv<<<BTn, 256, 0, stream>>>(hh, convw, conv);
  k_gemm<<<dim3(DFFn/GTN, BTn/GTM, 1), 256, 0, stream>>>(hh, ew1, nullptr, emid,
      BTn, DFFn, Dn, 0,0,0,0, 1.0f, 0, 1);             // fused GELU
  k_gemm<<<dim3(Dn/GTN, BTn/GTM, 1), 256, 0, stream>>>(emid, ew2, nullptr, eout,
      BTn, Dn, DFFn, 0,0,0,0, 1.0f, 0, 0);

  // attention tier (single head, d=1024): materialized scores
  k_gemm<<<dim3(Dn/GTN, BTn/GTM, 1), 256, 0, stream>>>(hh, atwq, nullptr, q2,
      BTn, Dn, Dn, 0,0,0,0, 1.0f, 0, 0);
  k_gemm<<<dim3(Dn/GTN, BTn/GTM, 1), 256, 0, stream>>>(hh, atwk, nullptr, k2,
      BTn, Dn, Dn, 0,0,0,0, 1.0f, 0, 0);
  k_gemm<<<dim3(Dn/GTN, BTn/GTM, 1), 256, 0, stream>>>(hh, atwv, nullptr, v2,
      BTn, Dn, Dn, 0,0,0,0, 1.0f, 0, 0);
  k_gemm<<<dim3(Tn/GTN, Tn/GTM, Bn), 256, 0, stream>>>(q2, k2, nullptr, S2,
      Tn, Tn, Dn, (long long)Tn*Dn, (long long)Tn*Dn, (long long)Tn*Tn, 0,
      0.03125f /* 1/sqrt(1024) */, 1 /* transB */, 0);
  k_softmax2<<<BTn, 256, 0, stream>>>(S2);
  k_gemm<<<dim3(Dn/GTN, Tn/GTM, Bn), 256, 0, stream>>>(S2, v2, nullptr, a2o,
      Tn, Dn, Tn, (long long)Tn*Tn, (long long)Tn*Dn, (long long)Tn*Dn, 0,
      1.0f, 0, 0);
  k_gemm<<<dim3(Dn/GTN, BTn/GTM, 1), 256, 0, stream>>>(a2o, atwo, nullptr, a2,
      BTn, Dn, Dn, 0,0,0,0, 1.0f, 0, 0);

  // --- combine + sink scaling ---
  k_combine<<<BTn, 256, 0, stream>>>(x2, conv, eout, a2, rwout, gate, sinks, xout);
}
